// LatticeCrf_61521111548030
// MI455X (gfx1250) — matrix operand order repaired and checked by reading, not compiled, not run on hardware
//
#include <hip/hip_runtime.h>
#include <hip/hip_bf16.h>

// LatticeCrf.log_prob for MI455X (gfx1250, wave32).
//
// Memory-bound: scores = 128 MiB (fits in the 192 MB L2), HBM floor ~5.5us
// @ 23.3 TB/s. The forward recursion is a log-semiring matrix chain
// (associative), so we expose parallelism with a blocked scan:
//   Stage 1: 64*16 = 1024 workgroups; each folds CH=32 lattice matrices into
//            one 32x32 log-space chunk matrix using f16 WMMA matmuls on
//            exponentiated, per-row-renormalized values (f32 accumulate).
//            Score tiles are double-buffered into LDS with CDNA5 async
//            global->LDS DMA (ASYNCcnt) when available.
//   Stage 2: 64 workgroups; chains the 16 chunk matrices per sequence with a
//            streaming logsumexp matvec, adds w_final, computes the numerator
//            gather, writes out[b] = log_num - log_denom.

#define Bn 64
#define Tn 512
#define Vn 32
#define CH 32            // lattice steps folded per chunk
#define NC (Tn / CH)     // 16 chunks per sequence

#if defined(__has_builtin)
#  if __has_builtin(__builtin_amdgcn_global_load_async_to_lds_b128) && \
      __has_builtin(__builtin_amdgcn_s_wait_asynccnt)
#    define HAVE_ASYNC_LDS 1
#  endif
#  if __has_builtin(__builtin_amdgcn_update_dpp)
#    define HAVE_DPP 1
#  endif
#endif

typedef __attribute__((ext_vector_type(16))) _Float16 v16h;
typedef __attribute__((ext_vector_type(8)))  float    v8f;
typedef __attribute__((ext_vector_type(4)))  int      v4i;

union HV {
    v16h v;
    _Float16 h[16];
    uint4 q[2];
};

// Max-reduce across each 16-lane DPP row. Rotations by 1,2,4,8 cover all 16
// offsets (every 0..15 is a subset sum), pure VALU instead of ds_bpermute.
__device__ __forceinline__ float rowmax16(float v) {
#ifdef HAVE_DPP
#define DPP_MAX_STEP(ctrl)                                                    \
    v = fmaxf(v, __int_as_float(__builtin_amdgcn_update_dpp(                  \
                     __float_as_int(v), __float_as_int(v), (ctrl), 0xf, 0xf,  \
                     true)))
    DPP_MAX_STEP(0x121);  // row_ror:1
    DPP_MAX_STEP(0x122);  // row_ror:2
    DPP_MAX_STEP(0x124);  // row_ror:4
    DPP_MAX_STEP(0x128);  // row_ror:8
#undef DPP_MAX_STEP
#else
#pragma unroll
    for (int msk = 1; msk < 16; msk <<= 1)
        v = fmaxf(v, __shfl_xor(v, msk, 32));
#endif
    return v;
}

#ifdef HAVE_ASYNC_LDS
// Fire the 8 async b128 DMAs that move one 4 KB score tile into a padded
// (stride-33) LDS buffer. Tracked by ASYNCcnt; in-order completion.
__device__ __forceinline__ void issue_tile_async(const float* gsrc,
                                                 float* lbuf, int lane) {
#pragma unroll
    for (int i = 0; i < 8; ++i) {
        const int f   = 4 * (lane + 32 * i);
        const int row = f >> 5;
        const int col = f & 31;
        __builtin_amdgcn_global_load_async_to_lds_b128(
            (__attribute__((address_space(1))) v4i*)(void*)(gsrc + f),
            (__attribute__((address_space(3))) v4i*)(void*)(lbuf + row * 33 + col),
            0, 0);
    }
}
#endif

// ---------------------------------------------------------------------------
// Stage 1: per-(b, chunk) product of CH exponentiated lattice matrices.
// Maintains P (row-normalized, f16 operands / f32 accum) and per-row log
// scale; emits logM[i][j] = rowlog[i] + log(P[i][j]) to workspace.
// ---------------------------------------------------------------------------
__global__ __launch_bounds__(32)
void crf_chunk_kernel(const float* __restrict__ scores,
                      const float* __restrict__ w_tx,
                      const float* __restrict__ w_dur,
                      float* __restrict__ chunkM) {
    __shared__ __align__(16) float    lds_raw[2][Vn * 33];  // raw scores, padded
    __shared__ __align__(16) _Float16 lds_p[Vn * 40];       // stride 40 halves = 80B
    __shared__ float                  lds_rowlog[Vn];

    const int lane = threadIdx.x;          // 0..31, one wave per block
    const int blk  = blockIdx.x;           // 0..Bn*NC-1
    const int b    = blk / NC;
    const int c    = blk % NC;
    const long long base = (long long)(b * Tn + c * CH) * (Vn * Vn);

    const int h = lane >> 4;               // lane half (K-group select)
    const int n = lane & 15;               // column within 16-wide tile
    const int m = lane & 15;               // row within A tile

    // (w_tx + w_dur) gathered once into registers, pre-arranged in the ISA
    // 16-bit B layout: lane element e covers K = 16*(e/8) + 8*half + (e%8), cols n / n+16.
    float wb0[16], wb1[16];
#pragma unroll
    for (int e = 0; e < 16; ++e) {
        const int K = 16 * (e >> 3) + 8 * h + (e & 7);
        wb0[e] = w_tx[K * Vn + n]      + w_dur[K * Vn + n];
        wb1[e] = w_tx[K * Vn + 16 + n] + w_dur[K * Vn + 16 + n];
    }
    lds_rowlog[lane] = 0.0f;

    // P = identity, pre-packed in the ISA 16-bit A layout:
    // lane element e -> K = 16*(e/8) + 8*half + (e%8).
    HV aT, aB;
#pragma unroll
    for (int e = 0; e < 16; ++e) {
        const int K = 16 * (e >> 3) + 8 * h + (e & 7);
        aT.h[e] = (K == m)      ? (_Float16)1.0f : (_Float16)0.0f;
        aB.h[e] = (K == m + 16) ? (_Float16)1.0f : (_Float16)0.0f;
    }

#ifdef HAVE_ASYNC_LDS
    issue_tile_async(scores + base, lds_raw[0], lane);   // prologue: tile 0
#endif

    for (int t = 0; t < CH; ++t) {
        const float* tile = scores + base + (long long)t * (Vn * Vn);

#ifdef HAVE_ASYNC_LDS
        float* cur = lds_raw[t & 1];
        float* nxt = lds_raw[(t & 1) ^ 1];
        __builtin_amdgcn_s_wait_asynccnt(0);  // tile t landed in LDS
        __syncthreads();                       // ordering point for LDS reads
        if (t + 1 < CH) issue_tile_async(tile + Vn * Vn, nxt, lane);
#else
        float* cur = lds_raw[0];
        if (t + 1 < CH) __builtin_prefetch(tile + Vn * Vn, 0, 1);
        float4 s4[8];
#pragma unroll
        for (int i = 0; i < 8; ++i) {
            const int f = 4 * (lane + 32 * i);
            s4[i] = *(const float4*)(tile + f);
        }
        __syncthreads();                       // prior readers done
#pragma unroll
        for (int i = 0; i < 8; ++i) {
            const int f   = 4 * (lane + 32 * i);
            const int row = f >> 5;
            const int col = f & 31;
            *(float4*)&cur[row * 33 + col] = s4[i];
        }
        __syncthreads();
#endif

        // B tiles (exp(score + wconst), f16): lane element e holds E[K][n].
        HV b0, b1;
#pragma unroll
        for (int e = 0; e < 16; ++e) {
            const int K = 16 * (e >> 3) + 8 * h + (e & 7);
            b0.h[e] = (_Float16)__expf(cur[K * 33 + n]      + wb0[e]);
            b1.h[e] = (_Float16)__expf(cur[K * 33 + 16 + n] + wb1[e]);
        }

        const v8f z = {};
        v8f d00 = __builtin_amdgcn_wmma_f32_16x16x32_f16(false, aT.v, false, b0.v, (short)0, z, false, false);
        v8f d01 = __builtin_amdgcn_wmma_f32_16x16x32_f16(false, aT.v, false, b1.v, (short)0, z, false, false);
        v8f d10 = __builtin_amdgcn_wmma_f32_16x16x32_f16(false, aB.v, false, b0.v, (short)0, z, false, false);
        v8f d11 = __builtin_amdgcn_wmma_f32_16x16x32_f16(false, aB.v, false, b1.v, (short)0, z, false, false);

        // Per-row max across 32 columns (D layout: VGPR r holds row 8*half+r,
        // its 16 columns live in one 16-lane half -> DPP row reduction).
        float rT[8], rB[8];
#pragma unroll
        for (int r = 0; r < 8; ++r) {
            rT[r] = rowmax16(fmaxf(d00[r], d01[r]));
            rB[r] = rowmax16(fmaxf(d10[r], d11[r]));
        }
        // One designated lane per row accumulates the log scale.
#pragma unroll
        for (int r = 0; r < 8; ++r) {
            if (n == r) {
                lds_rowlog[8 * h + r]      += __logf(rT[r]);
                lds_rowlog[16 + 8 * h + r] += __logf(rB[r]);
            }
        }
        // Normalize rows to max 1 and stash P as f16 (row-major, stride 40).
#pragma unroll
        for (int r = 0; r < 8; ++r) {
            const float iT = 1.0f / rT[r];
            const float iB = 1.0f / rB[r];
            lds_p[(8 * h + r) * 40 + n]           = (_Float16)(d00[r] * iT);
            lds_p[(8 * h + r) * 40 + 16 + n]      = (_Float16)(d01[r] * iT);
            lds_p[(16 + 8 * h + r) * 40 + n]      = (_Float16)(d10[r] * iB);
            lds_p[(16 + 8 * h + r) * 40 + 16 + n] = (_Float16)(d11[r] * iB);
        }
        __syncthreads();

        // Reload P in A layout: two contiguous 8-half runs per tile,
        // 16B-aligned thanks to the stride-40 (80B) row pitch.
        aT.q[0] = *(const uint4*)&lds_p[m * 40 + 8 * h];
        aT.q[1] = *(const uint4*)&lds_p[m * 40 + 16 + 8 * h];
        aB.q[0] = *(const uint4*)&lds_p[(m + 16) * 40 + 8 * h];
        aB.q[1] = *(const uint4*)&lds_p[(m + 16) * 40 + 16 + 8 * h];
        __syncthreads();
    }

    float* outM = chunkM + (long long)blk * (Vn * Vn);
    for (int idx = lane; idx < Vn * Vn; idx += 32) {
        const int i = idx >> 5;
        const int j = idx & 31;
        outM[idx] = lds_rowlog[i] + __logf((float)lds_p[i * 40 + j]);
    }
}

// ---------------------------------------------------------------------------
// Stage 2: chain chunk matrices per sequence (streaming logsumexp matvec),
// fold in w_final, compute the numerator gather, emit out[b].
// ---------------------------------------------------------------------------
__global__ __launch_bounds__(32)
void crf_finalize_kernel(const float* __restrict__ chunkM,
                         const float* __restrict__ scores,
                         const int*   __restrict__ targets,
                         const float* __restrict__ w_tx,
                         const float* __restrict__ w_init,
                         const float* __restrict__ w_final,
                         const float* __restrict__ w_dur,
                         float* __restrict__ out) {
    __shared__ float aSh[Vn];
    const int lane = threadIdx.x;   // state j
    const int b    = blockIdx.x;

    float alpha = w_init[lane];
    for (int c = 0; c < NC; ++c) {
        __syncthreads();
        aSh[lane] = alpha;
        __syncthreads();
        const float* M = chunkM + (long long)(b * NC + c) * (Vn * Vn);
        float mx = -INFINITY, s = 0.0f;
        for (int i = 0; i < Vn; ++i) {
            const float v  = aSh[i] + M[i * Vn + lane];
            const float nm = fmaxf(mx, v);
            if (nm > -INFINITY) {
                s  = s * __expf(mx - nm) + __expf(v - nm);
                mx = nm;
            }
        }
        alpha = mx + __logf(s);
    }

    // log_denom = logsumexp_j(alpha[j] + w_final[j]) across the wave.
    float v  = alpha + w_final[lane];
    float mx = v;
#pragma unroll
    for (int k = 1; k < 32; k <<= 1) mx = fmaxf(mx, __shfl_xor(mx, k, 32));
    float s = __expf(v - mx);
#pragma unroll
    for (int k = 1; k < 32; k <<= 1) s += __shfl_xor(s, k, 32);
    const float denom = mx + __logf(s);

    // Numerator: sum_t step[b,t,src,dst] + w_init[s0] + w_final[sT].
    float acc = 0.0f;
    for (int t = lane; t < Tn; t += 32) {
        const int src = targets[b * (Tn + 1) + t];
        const int dst = targets[b * (Tn + 1) + t + 1];
        acc += scores[(((long long)b * Tn + t) * Vn + src) * Vn + dst]
             + w_tx[src * Vn + dst] + w_dur[src * Vn + dst];
    }
#pragma unroll
    for (int k = 1; k < 32; k <<= 1) acc += __shfl_xor(acc, k, 32);

    const int s0 = targets[b * (Tn + 1)];
    const int sT = targets[b * (Tn + 1) + Tn];
    const float num = acc + w_init[s0] + w_final[sT];
    if (lane == 0) out[b] = num - denom;
}

extern "C" void kernel_launch(void* const* d_in, const int* in_sizes, int n_in,
                              void* d_out, int out_size, void* d_ws, size_t ws_size,
                              hipStream_t stream) {
    const float* scores  = (const float*)d_in[0];
    const int*   targets = (const int*)d_in[1];
    const float* w_tx    = (const float*)d_in[2];
    const float* w_init  = (const float*)d_in[3];
    const float* w_final = (const float*)d_in[4];
    const float* w_dur   = (const float*)d_in[5];
    float* out    = (float*)d_out;
    float* chunkM = (float*)d_ws;   // Bn*NC*Vn*Vn floats = 4 MiB, overwritten every call

    crf_chunk_kernel<<<Bn * NC, 32, 0, stream>>>(scores, w_tx, w_dur, chunkM);
    crf_finalize_kernel<<<Bn, 32, 0, stream>>>(chunkM, scores, targets,
                                               w_tx, w_init, w_final, w_dur, out);
}
